// AttnEmbed_33913061769233
// MI455X (gfx1250) — matrix operand order repaired and checked by reading, not compiled, not run on hardware
//
#include <hip/hip_runtime.h>
#include <hip/hip_bf16.h>

// ---------------------------------------------------------------------------
// Types & WMMA helpers (CDNA5 / gfx1250, wave32)
// ---------------------------------------------------------------------------
typedef __bf16 bf16;
typedef float  v8f  __attribute__((ext_vector_type(8)));
typedef bf16   v16bf __attribute__((ext_vector_type(16)));
typedef bf16   v8bf  __attribute__((ext_vector_type(8)));
typedef unsigned int v4u __attribute__((ext_vector_type(4)));
typedef int v8i __attribute__((ext_vector_type(8)));
typedef int v4i __attribute__((ext_vector_type(4)));

__device__ __forceinline__ v8f wmma_bf16(v16bf a, v16bf bm, v8f c) {
    // D = A(16x32 bf16) x B(32x16 bf16) + C(16x16 f32)
    return __builtin_amdgcn_wmma_f32_16x16x32_bf16(false, a, false, bm,
                                                   (short)0, c, false, false);
}

// A fragment (16x32, MxK) from a row-major [rows x ld] bf16 buffer.
// ISA 7.12.2: lanes 0-15 -> M=lane, K = {0..7,16..23}; lanes 16-31 -> M=lane-16,
// K = {8..15, 24..31}.
__device__ __forceinline__ v16bf load_a_frag(const bf16* buf, int ld,
                                             int row0, int k0, int lane) {
    const bf16* p = buf + (size_t)(row0 + (lane & 15)) * ld + k0 + ((lane & 16) ? 8 : 0);
    v8bf lo = *(const v8bf*)p;          // 16B aligned (k0 multiple of 32)
    v8bf hi = *(const v8bf*)(p + 16);
    return __builtin_shufflevector(lo, hi, 0,1,2,3,4,5,6,7,8,9,10,11,12,13,14,15);
}

// B fragment (32x16, KxN) from a buffer stored column-contiguous, i.e. an
// array bt[n][k] (row n = output column n, contiguous in k, row stride ldk).
// lanes 0-15: K = k0+{0..7,16..23}, lanes 16-31: K = k0+{8..15,24..31}; N = n0 + (lane&15).
__device__ __forceinline__ v16bf load_bt_frag(const bf16* bt, int ldk,
                                              int k0, int n0, int lane) {
    const bf16* p = bt + (size_t)(n0 + (lane & 15)) * ldk + k0 + ((lane & 16) ? 8 : 0);
    v8bf lo = *(const v8bf*)p;
    v8bf hi = *(const v8bf*)(p + 16);
    return __builtin_shufflevector(lo, hi, 0,1,2,3,4,5,6,7,8,9,10,11,12,13,14,15);
}

__device__ __forceinline__ unsigned int pack2bf(float a, float b) {
    unsigned short x = __builtin_bit_cast(unsigned short, (bf16)a);
    unsigned short y = __builtin_bit_cast(unsigned short, (bf16)b);
    return (unsigned int)x | ((unsigned int)y << 16);
}

// ---------------------------------------------------------------------------
// Tensor Data Mover: 2D bf16 tile (tile_d1 rows of tile_d0 elements, row
// stride stride0 elements) from global -> LDS, per CDNA5 ISA D# layout.
// Group0: count=1 | lds_addr | global_addr[56:0] | type=2
// Group1: data_size=1(2B) | tensor_dim0/1 | tile_dim0/1 | tensor_dim0_stride
// This toolchain's builtin takes 6 args (g0 v4u, g1 v8i, g2 v4i, g3 v4i,
// extra v8i, cpol i32) -- zero the unused trailing groups.
// ---------------------------------------------------------------------------
__device__ __forceinline__ void tdm_load_2d(const void* lds_dst, const void* gsrc,
                                            unsigned tile_d0, unsigned tile_d1,
                                            unsigned long long stride0) {
    unsigned long long ga = (unsigned long long)(uintptr_t)gsrc;
    unsigned lds_addr = (unsigned)(uintptr_t)lds_dst;   // LDS aperture: low 32 bits
    unsigned long long td0 = tile_d0, td1 = tile_d1;
    v4u g0;
    g0[0] = 1u;                                          // count=1, user D#
    g0[1] = lds_addr;
    g0[2] = (unsigned)(ga & 0xFFFFFFFFu);
    g0[3] = (unsigned)((ga >> 32) & 0x1FFFFFFu) | (2u << 30);   // type=2 "image"
    v8i g1;
    g1[0] = 0x10000;                                     // data_size=1 -> 2 bytes
    g1[1] = (int)((td0 & 0xFFFFu) << 16);                // tensor_dim0[15:0]
    g1[2] = (int)(((td0 >> 16) & 0xFFFFu) | ((td1 & 0xFFFFu) << 16));
    g1[3] = (int)(((td1 >> 16) & 0xFFFFu) | ((tile_d0 & 0xFFFFu) << 16));
    g1[4] = (int)(tile_d1 & 0xFFFFu);                    // tile_dim1 (tile_dim2=0)
    g1[5] = (int)(stride0 & 0xFFFFFFFFu);                // tensor_dim0_stride lo
    g1[6] = (int)((stride0 >> 32) & 0xFFFFu);            // stride hi (stride1=0)
    g1[7] = 0;
    v4i z4 = {0, 0, 0, 0};
    v8i z8 = {0, 0, 0, 0, 0, 0, 0, 0};
    __builtin_amdgcn_tensor_load_to_lds(g0, g1, z4, z4, z8, 0);
}

// ---------------------------------------------------------------------------
// ws byte offsets (all multiples of 256)
// ---------------------------------------------------------------------------
#define OFF_POS    ((size_t)0)            // 4096*256 f32   (4 MB)
#define OFF_KPOS   ((size_t)4194304)      // 4096*256 f32   (4 MB)
#define OFF_E3     ((size_t)8388608)      // 4*64*768 f32
#define OFF_EMBQ2  ((size_t)9175040)      // 4*64*256 f32
#define OFF_QF     ((size_t)9437184)      // 4*64*256 f32
#define OFF_KWBF   ((size_t)9699328)      // 256*256 bf16
#define OFF_VWBF   ((size_t)9830400)      // 256*256 bf16
#define OFF_L1BF   ((size_t)9961472)      // 1024*256 bf16
#define OFF_L2BF   ((size_t)10485760)     // 256*1024 bf16
#define OFF_RESP   ((size_t)11010048)     // 4*16*64*256 f32 (pooled sums, 16 MB)
#define OFF_RESF   ((size_t)27787264)     // 4*16*64*256 f32 (projected res, 16 MB)
#define OFF_KBF    ((size_t)44564480)     // 4*16*4096*256 bf16 (128 MB) [t][c]
#define OFF_VTBF   ((size_t)178782208)    // 4*16*256*4096 bf16 (128 MB) [c][t]

// ---------------------------------------------------------------------------
// 1) Positional-encoding table: pos[t][d], t = ty*64+tx
// ---------------------------------------------------------------------------
__global__ __launch_bounds__(256) void k_pos(float* __restrict__ pos) {
    int t = blockIdx.x;            // 0..4095
    int d = threadIdx.x;           // 0..255
    int ty = t >> 6, tx = t & 63;
    float v  = (d < 128) ? (float)(ty + 1) : (float)(tx + 1);
    int  dd  = d & 127;
    float freq = __powf(10000.f, (float)(dd & ~1) * (1.f / 128.f));
    float a = v / freq;
    pos[(size_t)t * 256 + d] = (dd & 1) ? __cosf(a) : __sinf(a);
}

// ---------------------------------------------------------------------------
// 2) fp32 -> bf16 weight conversion
// ---------------------------------------------------------------------------
__global__ __launch_bounds__(256) void k_cvt(const float* __restrict__ kw,
                                             const float* __restrict__ vw,
                                             const float* __restrict__ l1,
                                             const float* __restrict__ l2,
                                             bf16* kwbf, bf16* vwbf,
                                             bf16* l1bf, bf16* l2bf) {
    int i = blockIdx.x * 256 + threadIdx.x;     // grid 1024 -> i < 262144
    if (i < 65536) { kwbf[i] = (bf16)kw[i]; vwbf[i] = (bf16)vw[i]; }
    l1bf[i] = (bf16)l1[i];
    l2bf[i] = (bf16)l2[i];
}

// ---------------------------------------------------------------------------
// 3) kpos[t][n] = sum_k pos[t][k]*kw[n][k] + kw_b[n]   (folds pos into K bias)
// ---------------------------------------------------------------------------
__global__ __launch_bounds__(256) void k_kpos(const float* __restrict__ pos,
                                              const float* __restrict__ kw,
                                              const float* __restrict__ kb,
                                              float* __restrict__ kpos) {
    __shared__ float row[256];
    int t = blockIdx.x, n = threadIdx.x;
    row[n] = pos[(size_t)t * 256 + n];
    __syncthreads();
    float acc = kb[n];
    const float* wr = kw + (size_t)n * 256;
    #pragma unroll 8
    for (int k = 0; k < 256; ++k) acc += row[k] * wr[k];
    kpos[(size_t)t * 256 + n] = acc;
}

// ---------------------------------------------------------------------------
// 4) e3[b][t][0..767] = emb[b][t] @ embW^T + embW_b   (qe|ke|ve)
// ---------------------------------------------------------------------------
__global__ __launch_bounds__(256) void k_emb_gemm(const float* __restrict__ emb,
                                                  const float* __restrict__ W,
                                                  const float* __restrict__ bias,
                                                  float* __restrict__ e3) {
    __shared__ float row[256];
    int bt = blockIdx.x;               // b*64+t, grid 256
    int n  = threadIdx.x;
    row[n] = emb[(size_t)bt * 256 + n];
    __syncthreads();
    for (int jj = 0; jj < 3; ++jj) {
        int j = n + jj * 256;
        float acc = bias[j];
        const float* wr = W + (size_t)j * 256;
        #pragma unroll 8
        for (int k = 0; k < 256; ++k) acc += row[k] * wr[k];
        e3[(size_t)bt * 768 + j] = acc;
    }
}

// ---------------------------------------------------------------------------
// 5) emb self-attention:  embq2 = LN1(softmax(qe ke^T/16) ve + qe) + emb
// ---------------------------------------------------------------------------
__global__ __launch_bounds__(256) void k_emb_attn(const float* __restrict__ e3,
                                                  const float* __restrict__ emb,
                                                  const float* __restrict__ g,
                                                  const float* __restrict__ be,
                                                  float* __restrict__ embq2) {
    extern __shared__ char dyn_smem[];
    float* S = (float*)dyn_smem;                 // [64][64]
    float* O = (float*)(dyn_smem + 16384);       // [64][256]
    int b = blockIdx.x, tid = threadIdx.x;
    const float* e3b = e3 + (size_t)b * 64 * 768;
    for (int i = 0; i < 16; ++i) {
        int idx = tid + (i << 8);
        int qi = idx >> 6, kj = idx & 63;
        const float* qr = e3b + (size_t)qi * 768;
        const float* kr = e3b + (size_t)kj * 768 + 256;
        float acc = 0.f;
        #pragma unroll 8
        for (int k = 0; k < 256; ++k) acc += qr[k] * kr[k];
        S[qi * 64 + kj] = acc * 0.0625f;
    }
    __syncthreads();
    if (tid < 64) {
        float mx = -1e30f;
        for (int j = 0; j < 64; ++j) mx = fmaxf(mx, S[tid * 64 + j]);
        float sum = 0.f;
        for (int j = 0; j < 64; ++j) { float e = __expf(S[tid * 64 + j] - mx); S[tid * 64 + j] = e; sum += e; }
        float inv = 1.f / sum;
        for (int j = 0; j < 64; ++j) S[tid * 64 + j] *= inv;
    }
    __syncthreads();
    for (int i = 0; i < 64; ++i) {
        int idx = tid + (i << 8);
        int qi = idx >> 8, c = idx & 255;
        float acc = e3b[(size_t)qi * 768 + c];               // + qe
        for (int j = 0; j < 64; ++j) acc += S[qi * 64 + j] * e3b[(size_t)j * 768 + 512 + c];
        O[qi * 256 + c] = acc;
    }
    __syncthreads();
    if (tid < 64) {
        const float* r = O + tid * 256;
        float mu = 0.f;
        for (int c = 0; c < 256; ++c) mu += r[c];
        mu *= (1.f / 256.f);
        float var = 0.f;
        for (int c = 0; c < 256; ++c) { float d = r[c] - mu; var += d * d; }
        var *= (1.f / 256.f);
        float rstd = __frsqrt_rn(var + 1e-5f);
        for (int c = 0; c < 256; ++c)
            embq2[((size_t)b * 64 + tid) * 256 + c] =
                (r[c] - mu) * rstd * g[c] + be[c] + emb[((size_t)b * 64 + tid) * 256 + c];
    }
}

// ---------------------------------------------------------------------------
// 6) q[b][t][n] = embq2[b][t] @ qw^T + qw_b
// ---------------------------------------------------------------------------
__global__ __launch_bounds__(256) void k_emb_qproj(const float* __restrict__ embq2,
                                                   const float* __restrict__ qw,
                                                   const float* __restrict__ qb,
                                                   float* __restrict__ qf) {
    __shared__ float row[256];
    int bt = blockIdx.x, n = threadIdx.x;      // grid 256
    row[n] = embq2[(size_t)bt * 256 + n];
    __syncthreads();
    float acc = qb[n];
    const float* wr = qw + (size_t)n * 256;
    #pragma unroll 8
    for (int k = 0; k < 256; ++k) acc += row[k] * wr[k];
    qf[(size_t)bt * 256 + n] = acc;
}

// ---------------------------------------------------------------------------
// 7) K/V projection (WMMA) + res pooling.
//    One WG per (b, split, 8-row band) = 512 tokens. LDS: im tile bf16 256KB.
//    k[b][s][t][c]  (bf16, row-major, B-fragment-ready for Q@K^T)
//    vT[b][s][c][t] (bf16, transposed, B-fragment-ready for P@V)
// ---------------------------------------------------------------------------
__global__ __launch_bounds__(256) void k_kv_proj(const float* __restrict__ im,
                                                 const float* __restrict__ kpos,
                                                 const float* __restrict__ vwb,
                                                 const bf16* __restrict__ kwbf,
                                                 const bf16* __restrict__ vwbf,
                                                 bf16* __restrict__ kout,
                                                 bf16* __restrict__ vtout,
                                                 float* __restrict__ res_pool) {
    extern __shared__ char dyn_smem[];
    bf16* imbf = (bf16*)dyn_smem;                       // [512][256]
    int wg = blockIdx.x;                                // 0..511
    int b  = wg >> 7;
    int sp = (wg >> 3) & 15;
    int by = wg & 7;
    int sy = sp >> 2, sx = sp & 3;
    int tid = threadIdx.x, w = tid >> 5, lane = tid & 31;
    int y0 = sy * 64 + by * 8, x0 = sx * 64;
    const float* imB = im + (size_t)b * 256 * 65536;    // im[b][c][y][x]

    // --- load 512x256 im tile (coalesced along x), convert to bf16 in LDS ---
    for (int i = 0; i < 128; ++i) {
        int idx = tid + (i << 8);                       // 0..32767
        int c   = idx >> 7;
        int rem = idx & 127;
        int yy  = rem >> 4;
        int tx4 = (rem & 15) << 2;
        const float4 v4 = *(const float4*)(imB + (size_t)c * 65536 + (size_t)(y0 + yy) * 256 + x0 + tx4);
        bf16* d = imbf + (size_t)(yy * 64 + tx4) * 256 + c;
        d[0]   = (bf16)v4.x;
        d[256] = (bf16)v4.y;
        d[512] = (bf16)v4.z;
        d[768] = (bf16)v4.w;
    }
    __syncthreads();

    // --- res pooling: this band is exactly block-row `by` (8y x 8x sums) ---
    for (int i = 0; i < 8; ++i) {
        int o = tid + (i << 8);                         // 0..2047
        int bx = o >> 8, c = o & 255;
        float acc = 0.f;
        for (int yy = 0; yy < 8; ++yy) {
            const bf16* rp = imbf + (size_t)(yy * 64 + bx * 8) * 256 + c;
            #pragma unroll
            for (int xx = 0; xx < 8; ++xx) acc += (float)rp[(size_t)xx * 256];
        }
        res_pool[((((size_t)b * 16 + sp) * 8 + by) * 8 + bx) * 256 + c] = acc;
    }

    // --- projections: each wave owns 64 rows (4 M-frags), loops 16 N-tiles ---
    int T0 = by * 512;
    size_t bs = (size_t)(b * 16 + sp);
    int hi8 = (lane & 16) ? 8 : 0;
    for (int proj = 0; proj < 2; ++proj) {
        const bf16* W = proj ? vwbf : kwbf;             // [n][k] row-major
        for (int nt = 0; nt < 16; ++nt) {
            int n0 = nt * 16;
            int n  = n0 + (lane & 15);
            v8f acc[4];
            #pragma unroll
            for (int mi = 0; mi < 4; ++mi)
                #pragma unroll
                for (int r = 0; r < 8; ++r) acc[mi][r] = 0.f;
            #pragma unroll
            for (int kt = 0; kt < 8; ++kt) {
                v16bf bfrag = load_bt_frag(W, 256, kt * 32, n0, lane);
                #pragma unroll
                for (int mi = 0; mi < 4; ++mi) {
                    v16bf afrag = load_a_frag(imbf, 256, w * 64 + mi * 16, kt * 32, lane);
                    acc[mi] = wmma_bf16(afrag, bfrag, acc[mi]);
                }
            }
            if (proj == 0) {                            // K = im@kw^T + kpos
                #pragma unroll
                for (int mi = 0; mi < 4; ++mi) {
                    int mb = w * 64 + mi * 16 + hi8;
                    #pragma unroll
                    for (int r = 0; r < 8; ++r) {
                        int T = T0 + mb + r;
                        float val = acc[mi][r] + kpos[(size_t)T * 256 + n];
                        kout[(bs * 4096 + T) * 256 + n] = (bf16)val;
                    }
                }
            } else {                                    // V = im@vw^T + vw_b (transposed store)
                float bias = vwb[n];
                #pragma unroll
                for (int mi = 0; mi < 4; ++mi) {
                    int mb = w * 64 + mi * 16 + hi8;
                    uint4 u;
                    u.x = pack2bf(acc[mi][0] + bias, acc[mi][1] + bias);
                    u.y = pack2bf(acc[mi][2] + bias, acc[mi][3] + bias);
                    u.z = pack2bf(acc[mi][4] + bias, acc[mi][5] + bias);
                    u.w = pack2bf(acc[mi][6] + bias, acc[mi][7] + bias);
                    *(uint4*)(vtout + (bs * 256 + n) * 4096 + T0 + mb) = u;
                }
            }
        }
    }
}

// ---------------------------------------------------------------------------
// 8) res projection: resf[b][s][m][n] = (pool[m]/64) @ resup^T + resup_b
// ---------------------------------------------------------------------------
__global__ __launch_bounds__(256) void k_res_proj(const float* __restrict__ res_pool,
                                                  const float* __restrict__ W,
                                                  const float* __restrict__ bias,
                                                  float* __restrict__ resf) {
    __shared__ float row[256];
    int bm = blockIdx.x, n = threadIdx.x;       // grid 4096
    row[n] = res_pool[(size_t)bm * 256 + n] * (1.f / 64.f);
    __syncthreads();
    float acc = bias[n];
    const float* wr = W + (size_t)n * 256;
    #pragma unroll 8
    for (int k = 0; k < 256; ++k) acc += row[k] * wr[k];
    resf[(size_t)bm * 256 + n] = acc;
}

// ---------------------------------------------------------------------------
// 9) Fused flash attention + LN1 + res + MLP(gelu) + LN2  (WMMA + TDM)
//    One WG per (b, split), 8 waves.  Wave (om = w&3, on = w>>2) owns output
//    rows 16*om, cols 128*on.  K/V chunks staged into LDS by the Tensor Data
//    Mover, double-buffered, overlapped with WMMA compute.
//    Dynamic LDS map (bytes):
//      0      qbf [64][256] bf16 (phase1) / abf (phase2)
//      32768  K chunk buf0 [64][256] bf16 } phase1; hbf [64][1024] bf16 in
//      65536  V chunk buf0 [256][64] bf16 } phase2 spans 32768..163840
//      98304  K chunk buf1
//      131072 V chunk buf1
//      163840 att [64][256] f32 (both phases)
//      229376 Sbuf [64][64] f32 (phase1)
//      245760 pbf [64][64] bf16 (phase1)
//      253952 rowm/rowl/rowc [64]x3 f32 (phase1)
// ---------------------------------------------------------------------------
__global__ __launch_bounds__(256) void k_attn_mlp(const float* __restrict__ qf,
                                                  const bf16* __restrict__ kbf,
                                                  const bf16* __restrict__ vtbf,
                                                  const float* __restrict__ resf,
                                                  const float* __restrict__ n1g,
                                                  const float* __restrict__ n1b,
                                                  const float* __restrict__ png,
                                                  const float* __restrict__ pnb,
                                                  const bf16* __restrict__ l1bf,
                                                  const float* __restrict__ l1b,
                                                  const bf16* __restrict__ l2bf,
                                                  const float* __restrict__ l2b,
                                                  float* __restrict__ out) {
    extern __shared__ char dyn_smem[];
    bf16*  qbf  = (bf16*)dyn_smem;
    bf16*  kbuf0 = (bf16*)(dyn_smem + 32768);
    bf16*  vbuf0 = (bf16*)(dyn_smem + 65536);
    bf16*  kbuf1 = (bf16*)(dyn_smem + 98304);
    bf16*  vbuf1 = (bf16*)(dyn_smem + 131072);
    float* att  = (float*)(dyn_smem + 163840);
    float* Sbuf = (float*)(dyn_smem + 229376);
    bf16*  pbf  = (bf16*)(dyn_smem + 245760);
    float* rowm = (float*)(dyn_smem + 253952);
    float* rowl = rowm + 64;
    float* rowc = rowm + 128;
    bf16*  abf  = (bf16*)dyn_smem;                    // phase 2
    bf16*  hbf  = (bf16*)(dyn_smem + 32768);          // phase 2, 128KB

    int blk = blockIdx.x;                 // 0..63
    int b   = blk >> 4;
    int sp  = blk & 15;
    int tid = threadIdx.x, w = tid >> 5, lane = tid & 31;
    int om = w & 3, on = w >> 2;
    int hi8 = (lane & 16) ? 8 : 0;
    size_t bs = (size_t)(b * 16 + sp);
    const bf16* kbase  = kbf  + bs * 4096 * 256;      // [t][c]
    const bf16* vtbase = vtbf + bs * 256 * 4096;      // [c][t]

    // TDM stage of one 64-token chunk: K tile 64x256 contiguous, V tile
    // 256 rows x 64 elems with row stride 4096 (both bf16).
    auto stage = [&](int kc) {
        int bi = kc & 1;
        tdm_load_2d(bi ? kbuf1 : kbuf0, kbase + (size_t)kc * 64 * 256,
                    256, 64, 256);
        tdm_load_2d(bi ? vbuf1 : vbuf0, vtbase + (size_t)kc * 64,
                    64, 256, 4096);
    };

    // stage q (bf16) + init softmax state
    for (int i = tid; i < 64 * 256; i += 256) qbf[i] = (bf16)qf[(size_t)b * 64 * 256 + i];
    if (tid < 64) { rowm[tid] = -1e30f; rowl[tid] = 0.f; }
    if (w == 0) {
        stage(0);
        __builtin_amdgcn_s_wait_tensorcnt(0);
    }
    __syncthreads();

    v8f acc[8];
    #pragma unroll
    for (int nn = 0; nn < 8; ++nn)
        #pragma unroll
        for (int r = 0; r < 8; ++r) acc[nn][r] = 0.f;

    // ---- flash loop over 64 key chunks of 64 tokens ----
    for (int kc = 0; kc < 64; ++kc) {
        const bf16* kch = (kc & 1) ? kbuf1 : kbuf0;
        const bf16* vch = (kc & 1) ? vbuf1 : vbuf0;
        if (w == 0 && kc + 1 < 64) stage(kc + 1);      // async: overlaps compute below
        // scores tile: rows 16*om, cols 32*on .. +31
        {
            int sn0 = 32 * on;
            v8f s0, s1;
            #pragma unroll
            for (int r = 0; r < 8; ++r) { s0[r] = 0.f; s1[r] = 0.f; }
            #pragma unroll
            for (int kt = 0; kt < 8; ++kt) {
                v16bf a  = load_a_frag(qbf, 256, 16 * om, 32 * kt, lane);
                v16bf b0 = load_bt_frag(kch, 256, 32 * kt, sn0, lane);
                v16bf b1 = load_bt_frag(kch, 256, 32 * kt, sn0 + 16, lane);
                s0 = wmma_bf16(a, b0, s0);
                s1 = wmma_bf16(a, b1, s1);
            }
            int nS = lane & 15;
            #pragma unroll
            for (int r = 0; r < 8; ++r) {
                int m = 16 * om + r + hi8;
                Sbuf[m * 64 + sn0 + nS]      = s0[r] * 0.0625f;
                Sbuf[m * 64 + sn0 + 16 + nS] = s1[r] * 0.0625f;
            }
        }
        __syncthreads();
        // online softmax: waves on==0, lanes 0..15, one row each
        if (on == 0 && lane < 16) {
            int row = 16 * om + lane;
            float mo = rowm[row];
            float mx = mo;
            for (int j = 0; j < 64; ++j) mx = fmaxf(mx, Sbuf[row * 64 + j]);
            float corr = __expf(mo - mx);
            float l = rowl[row] * corr;
            for (int j = 0; j < 64; ++j) {
                float e = __expf(Sbuf[row * 64 + j] - mx);
                l += e;
                pbf[row * 64 + j] = (bf16)e;
            }
            rowm[row] = mx; rowl[row] = l; rowc[row] = corr;
        }
        __syncthreads();
        // rescale accumulator, then P@V (B frags from LDS V chunk, ldk=64)
        float cr[8];
        #pragma unroll
        for (int r = 0; r < 8; ++r) cr[r] = rowc[16 * om + r + hi8];
        #pragma unroll
        for (int nn = 0; nn < 8; ++nn)
            #pragma unroll
            for (int r = 0; r < 8; ++r) acc[nn][r] *= cr[r];
        v16bf a0 = load_a_frag(pbf, 64, 16 * om, 0, lane);
        v16bf a1 = load_a_frag(pbf, 64, 16 * om, 32, lane);
        #pragma unroll
        for (int nn = 0; nn < 8; ++nn) {
            int n0 = 128 * on + 16 * nn;
            v16bf b0 = load_bt_frag(vch, 64, 0,  n0, lane);
            v16bf b1 = load_bt_frag(vch, 64, 32, n0, lane);
            acc[nn] = wmma_bf16(a0, b0, acc[nn]);
            acc[nn] = wmma_bf16(a1, b1, acc[nn]);
        }
        // wait for next chunk's TDM transfers, then release buffers
        if (w == 0) __builtin_amdgcn_s_wait_tensorcnt(0);
        __syncthreads();
    }

    // ---- epilogue: normalize, +q, stage f32 ----
    float linv[8];
    #pragma unroll
    for (int r = 0; r < 8; ++r) linv[r] = 1.f / rowl[16 * om + r + hi8];
    int nO = lane & 15;
    #pragma unroll
    for (int nn = 0; nn < 8; ++nn) {
        int n = 128 * on + 16 * nn + nO;
        #pragma unroll
        for (int r = 0; r < 8; ++r) {
            int m = 16 * om + r + hi8;
            att[m * 256 + n] = acc[nn][r] * linv[r] + qf[((size_t)b * 64 + m) * 256 + n];
        }
    }
    __syncthreads();
    // LN1 + res ; produce bf16 activations for the MLP
    if (tid < 64) {
        float* r = att + tid * 256;
        float mu = 0.f;
        for (int c = 0; c < 256; ++c) mu += r[c];
        mu *= (1.f / 256.f);
        float var = 0.f;
        for (int c = 0; c < 256; ++c) { float d = r[c] - mu; var += d * d; }
        var *= (1.f / 256.f);
        float rstd = __frsqrt_rn(var + 1e-5f);
        const float* rr = resf + (bs * 64 + tid) * 256;
        for (int c = 0; c < 256; ++c) {
            float v = (r[c] - mu) * rstd * n1g[c] + n1b[c] + rr[c];
            r[c] = v;
            abf[tid * 256 + c] = (bf16)v;
        }
    }
    __syncthreads();

    // ---- MLP GEMM 1: [64x256]@[256x1024], exact GELU ----
    for (int nt = 0; nt < 32; ++nt) {
        int n0 = 512 * on + 16 * nt;
        int n  = n0 + (lane & 15);
        float bias = l1b[n];
        v8f a1c;
        #pragma unroll
        for (int r = 0; r < 8; ++r) a1c[r] = bias;
        #pragma unroll
        for (int kt = 0; kt < 8; ++kt) {
            v16bf a = load_a_frag(abf, 256, 16 * om, 32 * kt, lane);
            v16bf bm = load_bt_frag(l1bf, 256, 32 * kt, n0, lane);
            a1c = wmma_bf16(a, bm, a1c);
        }
        #pragma unroll
        for (int r = 0; r < 8; ++r) {
            float x = a1c[r];
            float gl = 0.5f * x * (1.f + erff(x * 0.70710678118654752f));
            hbf[(size_t)(16 * om + r + hi8) * 1024 + n] = (bf16)gl;
        }
    }
    __syncthreads();

    // ---- MLP GEMM 2: [64x1024]@[1024x256] + residual ----
    for (int nt = 0; nt < 8; ++nt) {
        int n0 = 128 * on + 16 * nt;
        int n  = n0 + (lane & 15);
        float bias = l2b[n];
        v8f a2c;
        #pragma unroll
        for (int r = 0; r < 8; ++r) a2c[r] = bias;
        #pragma unroll
        for (int kt = 0; kt < 32; ++kt) {
            v16bf a = load_a_frag(hbf, 1024, 16 * om, 32 * kt, lane);
            v16bf bm = load_bt_frag(l2bf, 1024, 32 * kt, n0, lane);
            a2c = wmma_bf16(a, bm, a2c);
        }
        #pragma unroll
        for (int r = 0; r < 8; ++r) {
            int m = 16 * om + r + hi8;
            att[m * 256 + n] = a2c[r] + att[m * 256 + n];
        }
    }
    __syncthreads();

    // ---- LN2 + store ----
    if (tid < 64) {
        const float* r = att + tid * 256;
        float mu = 0.f;
        for (int c = 0; c < 256; ++c) mu += r[c];
        mu *= (1.f / 256.f);
        float var = 0.f;
        for (int c = 0; c < 256; ++c) { float d = r[c] - mu; var += d * d; }
        var *= (1.f / 256.f);
        float rstd = __frsqrt_rn(var + 1e-5f);
        float* op = out + ((size_t)b * 1024 + sp * 64 + tid) * 256;
        for (int c = 0; c < 256; ++c)
            op[c] = (r[c] - mu) * rstd * png[c] + pnb[c];
    }
}

// ---------------------------------------------------------------------------
// Host launcher
// ---------------------------------------------------------------------------
extern "C" void kernel_launch(void* const* d_in, const int* in_sizes, int n_in,
                              void* d_out, int out_size, void* d_ws, size_t ws_size,
                              hipStream_t stream) {
    const float* im      = (const float*)d_in[0];
    const float* emb     = (const float*)d_in[1];
    const float* qw_w    = (const float*)d_in[2];
    const float* qw_b    = (const float*)d_in[3];
    const float* kw_w    = (const float*)d_in[4];
    const float* kw_b    = (const float*)d_in[5];
    const float* vw_w    = (const float*)d_in[6];
    const float* vw_b    = (const float*)d_in[7];
    const float* embW_w  = (const float*)d_in[8];
    const float* embW_b  = (const float*)d_in[9];
    const float* n1g     = (const float*)d_in[10];
    const float* n1b     = (const float*)d_in[11];
    const float* png     = (const float*)d_in[12];
    const float* pnb     = (const float*)d_in[13];
    const float* lin1_w  = (const float*)d_in[14];
    const float* lin1_b  = (const float*)d_in[15];
    const float* lin2_w  = (const float*)d_in[16];
    const float* lin2_b  = (const float*)d_in[17];
    const float* resup_w = (const float*)d_in[18];
    const float* resup_b = (const float*)d_in[19];
    float* outp = (float*)d_out;
    char*  ws   = (char*)d_ws;

    float* pos   = (float*)(ws + OFF_POS);
    float* kpos  = (float*)(ws + OFF_KPOS);
    float* e3    = (float*)(ws + OFF_E3);
    float* embq2 = (float*)(ws + OFF_EMBQ2);
    float* qfp   = (float*)(ws + OFF_QF);
    bf16*  kwbf  = (bf16*)(ws + OFF_KWBF);
    bf16*  vwbf  = (bf16*)(ws + OFF_VWBF);
    bf16*  l1bf  = (bf16*)(ws + OFF_L1BF);
    bf16*  l2bf  = (bf16*)(ws + OFF_L2BF);
    float* respl = (float*)(ws + OFF_RESP);
    float* resfp = (float*)(ws + OFF_RESF);
    bf16*  kbfp  = (bf16*)(ws + OFF_KBF);
    bf16*  vtbfp = (bf16*)(ws + OFF_VTBF);

    k_pos<<<4096, 256, 0, stream>>>(pos);
    k_cvt<<<1024, 256, 0, stream>>>(kw_w, vw_w, lin1_w, lin2_w, kwbf, vwbf, l1bf, l2bf);
    k_kpos<<<4096, 256, 0, stream>>>(pos, kw_w, kw_b, kpos);
    k_emb_gemm<<<256, 256, 0, stream>>>(emb, embW_w, embW_b, e3);
    k_emb_attn<<<4, 256, 81920, stream>>>(e3, emb, n1g, n1b, embq2);
    k_emb_qproj<<<256, 256, 0, stream>>>(embq2, qw_w, qw_b, qfp);
    k_kv_proj<<<512, 256, 262144, stream>>>(im, kpos, vw_b, kwbf, vwbf, kbfp, vtbfp, respl);
    k_res_proj<<<4096, 256, 0, stream>>>(respl, resup_w, resup_b, resfp);
    k_attn_mlp<<<64, 256, 254976, stream>>>(qfp, kbfp, vtbfp, resfp, n1g, n1b, png, pnb,
                                            l1bf, lin1_b, l2bf, lin2_b, outp);
    (void)in_sizes; (void)n_in; (void)out_size; (void)ws_size;
}
